// DualHeadModel_24618752541197
// MI455X (gfx1250) — compile-verified, matrix operand order checked by reading, not run on hardware
//
#include <hip/hip_runtime.h>
#include <hip/hip_bf16.h>

// ---------------------------------------------------------------------------
// DualHeadModel for MI455X (gfx1250): SAGE convs + dual edge heads + dual
// BatchNorm MLPs.  GEMMs: v_wmma_f32_16x16x32_bf16, B staged to LDS via
// global_load_async_to_lds_b128 (ASYNCcnt) with double buffering.
// ---------------------------------------------------------------------------

typedef __attribute__((ext_vector_type(16))) __bf16 v16bf;
typedef __attribute__((ext_vector_type(8)))  float  v8f;

#define N_NODES 50000
#define IN_C    256
#define HID     256
#define E_MSG   800000
#define E_SUP   100000
#define BN_EPS  1e-5f

// d_in flattening (dict insertion order, recursively):
#define IN_X    0
#define IN_SUP  1
#define IN_MSG  2
#define IN_MSGW 3
#define IN_W1N  4
#define IN_B1   5
#define IN_W1R  6
#define IN_W2N  7
#define IN_B2   8
#define IN_W2R  9
#define IN_EPW  10
#define IN_EPB  11
#define IN_EWW  12
#define IN_EWB  13
#define IN_PROB 14   // L0:{W,b,g,be} L1:{..} L2:{..} L3:{W,b} -> 14 tensors
#define IN_WT   28

// ---------------------------------------------------------------------------
// utility kernels
// ---------------------------------------------------------------------------
__global__ void k_zero_f32(float* p, size_t n) {
  size_t i = (size_t)blockIdx.x * blockDim.x + threadIdx.x;
  if (i < n) p[i] = 0.0f;
}

__global__ void k_f32_to_bf16(const float* __restrict__ in, __bf16* __restrict__ out, size_t n) {
  size_t i = (size_t)blockIdx.x * blockDim.x + threadIdx.x;
  if (i < n) out[i] = (__bf16)in[i];
}

// W [K x N] row-major f32  ->  Wt [N x K] row-major bf16
__global__ void k_wt_convert(const float* __restrict__ W, __bf16* __restrict__ Wt, int K, int Nn) {
  size_t i = (size_t)blockIdx.x * blockDim.x + threadIdx.x;
  if (i >= (size_t)K * Nn) return;
  int n = (int)(i % Nn);
  int k = (int)(i / Nn);
  Wt[(size_t)n * K + k] = (__bf16)W[i];
}

// ---------------------------------------------------------------------------
// SAGE conv pieces
// ---------------------------------------------------------------------------
__global__ void k_sage_cnt(const int* __restrict__ me, float* __restrict__ cnt, int E) {
  int e = blockIdx.x * blockDim.x + threadIdx.x;
  if (e >= E) return;
  atomicAdd(&cnt[me[E + e]], 1.0f);   // dst row
}

__global__ void k_sage_msg(const float* __restrict__ x, const int* __restrict__ me,
                           const float* __restrict__ w, float* __restrict__ agg,
                           int E, int d) {
  size_t i = (size_t)blockIdx.x * blockDim.x + threadIdx.x;
  if (i >= (size_t)E * d) return;
  int e = (int)(i / d);
  int c = (int)(i % d);
  int s = me[e];
  int t = me[E + e];
  atomicAdd(&agg[(size_t)t * d + c], x[(size_t)s * d + c] * w[e]);
}

// agg /= max(cnt,1), emit bf16
__global__ void k_sage_fin(const float* __restrict__ agg, const float* __restrict__ cnt,
                           __bf16* __restrict__ aggb, int Nn, int d) {
  size_t i = (size_t)blockIdx.x * blockDim.x + threadIdx.x;
  if (i >= (size_t)Nn * d) return;
  int n = (int)(i / d);
  float c = fmaxf(cnt[n], 1.0f);
  aggb[i] = (__bf16)(agg[i] / c);
}

// ---------------------------------------------------------------------------
// WMMA GEMM: C[M x N] = A[M x K](bf16,row) * Bt[N x K](bf16,row) (+bias)(+C)
// Block = 4 waves, each wave owns a 16x128 strip of C (8 accumulators) over
// adjacent M-tiles sharing one 128-col N-strip.  Wave 0 stages the 128x32 B
// tile into double-buffered LDS with global_load_async_to_lds_b128; all waves
// read fragments back via ds_load_b128.  M%16==0, N%128==0, K%32==0.
// ---------------------------------------------------------------------------
template <bool OUT_BF16, bool ACCUM>
__launch_bounds__(128)
__global__ void k_gemm_wmma(const __bf16* __restrict__ A, const __bf16* __restrict__ Bt,
                            void* __restrict__ C, const float* __restrict__ bias,
                            int M, int N, int K) {
  const int NT = 8;                         // 16 x 128 per wave
  __shared__ uint4 smemB[2][512];           // 2 x 8KB: 128 cols x 32 k of bf16

  int lane = threadIdx.x & 31;
  int wave = threadIdx.x >> 5;
  int nstrips = N / 128;
  int bm = blockIdx.x / nstrips;
  int ns = blockIdx.x % nstrips;
  int mt = bm * 4 + wave;
  int mtiles = M / 16;
  if (mt >= mtiles) return;                 // whole tail waves exit (barrier-safe)

  int half = lane >> 4;                     // K-half select per ISA 16-bit layout
  int r    = lane & 15;                     // row (A) / column (B)
  int col0 = ns * 128;

  const __bf16* aptr = A + (size_t)(mt * 16 + r) * K;
  const char*   bbase = (const char*)Bt;
  uint32_t lds0 = (uint32_t)(uintptr_t)&smemB[0][0];

  // wave 0 stages a 128x32 B tile (8KB = 512 16B chunks, 16 per lane)
  auto issueB = [&](int buf, int kb) {
    uint32_t lb = lds0 + (uint32_t)buf * 8192u;
#pragma unroll
    for (int i = 0; i < 16; ++i) {
      int chunk = lane + i * 32;            // 0..511
      int c = chunk >> 2, sub = chunk & 3;  // column, 16B sub-chunk
      uint64_t ga = (uint64_t)(uintptr_t)(bbase +
                     ((size_t)(col0 + c) * K + kb) * 2 + (size_t)sub * 16);
      uint32_t la = lb + (uint32_t)chunk * 16u;
      asm volatile("global_load_async_to_lds_b128 %0, %1, off"
                   :: "v"(la), "v"(ga) : "memory");
    }
  };

  v8f acc[NT];
#pragma unroll
  for (int t = 0; t < NT; ++t) acc[t] = (v8f){0,0,0,0,0,0,0,0};

  if (wave == 0) issueB(0, 0);
  int buf = 0;
  for (int kb = 0; kb < K; kb += 32) {
    if (wave == 0) asm volatile("s_wait_asynccnt 0" ::: "memory");
    __syncthreads();                                        // tile[buf] visible
    if (wave == 0 && kb + 32 < K) issueB(buf ^ 1, kb + 32); // overlap next tile

    union { uint4 u[2]; v16bf v; } au;
    au.u[0] = *(const uint4*)(aptr + kb + 8 * half);        // K = kb+8h..+7
    au.u[1] = *(const uint4*)(aptr + kb + 16 + 8 * half);   // K = kb+16+8h..+7

    const char* bcur = (const char*)&smemB[buf][0];
#pragma unroll
    for (int t = 0; t < NT; ++t) {
      const char* bcol = bcur + (size_t)(t * 16 + r) * 64;  // 64B column frag
      union { uint4 u[2]; v16bf v; } bu;
      bu.u[0] = *(const uint4*)(bcol + 16 * half);
      bu.u[1] = *(const uint4*)(bcol + 32 + 16 * half);
      acc[t] = __builtin_amdgcn_wmma_f32_16x16x32_bf16(
          false, au.v, false, bu.v, (short)0, acc[t], false, false);
    }
    buf ^= 1;
  }

#pragma unroll
  for (int t = 0; t < NT; ++t) {
    int col = col0 + t * 16 + r;
    float bval = (bias != nullptr) ? bias[col] : 0.0f;
#pragma unroll
    for (int j = 0; j < 8; ++j) {
      int row = mt * 16 + j + 8 * half;     // C/D: vgpr j -> M=j (+8 hi half)
      size_t idx = (size_t)row * N + col;
      float v = acc[t][j] + bval;
      if (OUT_BF16) {
        ((__bf16*)C)[idx] = (__bf16)v;
      } else {
        float* Cf = (float*)C;
        if (ACCUM) v += Cf[idx];
        Cf[idx] = v;
      }
    }
  }
}

// ---------------------------------------------------------------------------
// BatchNorm (training stats, biased var) over bf16 activations
// ---------------------------------------------------------------------------
__global__ void k_bn_stats(const __bf16* __restrict__ act, float* __restrict__ colsum,
                           float* __restrict__ colsq, int M, int C, int rowsPerBlk) {
  int c = blockIdx.x * blockDim.x + threadIdx.x;
  if (c >= C) return;
  int r0 = blockIdx.y * rowsPerBlk;
  int r1 = r0 + rowsPerBlk; if (r1 > M) r1 = M;
  float s = 0.0f, s2 = 0.0f;
  for (int r = r0; r < r1; ++r) {
    float v = (float)act[(size_t)r * C + c];
    s += v; s2 += v * v;
  }
  atomicAdd(&colsum[c], s);
  atomicAdd(&colsq[c], s2);
}

__global__ void k_bn_apply(__bf16* __restrict__ act, const float* __restrict__ colsum,
                           const float* __restrict__ colsq, const float* __restrict__ g,
                           const float* __restrict__ be, int M, int C) {
  size_t i = (size_t)blockIdx.x * blockDim.x + threadIdx.x;
  if (i >= (size_t)M * C) return;
  int c = (int)(i % C);
  float inv_m = 1.0f / (float)M;
  float mu  = colsum[c] * inv_m;
  float var = colsq[c] * inv_m - mu * mu;
  float v = ((float)act[i] - mu) * rsqrtf(var + BN_EPS) * g[c] + be[c];
  act[i] = (__bf16)fmaxf(v, 0.0f);
}

// ---------------------------------------------------------------------------
// edge features from raw x -> bf16 [E_SUP x 512]  (concat [a | p])
// ---------------------------------------------------------------------------
__global__ void k_build_nf(const float* __restrict__ x, const int* __restrict__ se,
                           __bf16* __restrict__ nf, int E, int d) {
  size_t i = (size_t)blockIdx.x * blockDim.x + threadIdx.x;
  if (i >= (size_t)E * d) return;
  int e = (int)(i / d);
  int c = (int)(i % d);
  int s = se[e];
  int t = se[E + e];
  float xs = x[(size_t)s * d + c];
  float xt = x[(size_t)t * d + c];
  nf[(size_t)e * (2 * d) + c]     = (__bf16)(xs + xt);
  nf[(size_t)e * (2 * d) + d + c] = (__bf16)(xs * xt);
}

// ---------------------------------------------------------------------------
// GNN edge heads: ef = [h_s+h_t | h_s*h_t]; ew = relu(ef.w_ew+b); ep = ef.w_ep+b
// ---------------------------------------------------------------------------
__global__ void k_gnn_edge_head(const float* __restrict__ h, const int* __restrict__ se,
                                const float* __restrict__ ew_w, const float* __restrict__ ew_b,
                                const float* __restrict__ ep_w, const float* __restrict__ ep_b,
                                float* __restrict__ out_ew, float* __restrict__ out_ep,
                                int E, int d) {
  int e = blockIdx.x * blockDim.x + threadIdx.x;
  if (e >= E) return;
  int s = se[e];
  int t = se[E + e];
  const float* hs = h + (size_t)s * d;
  const float* ht = h + (size_t)t * d;
  float ew = 0.0f, ep = 0.0f;
  for (int c = 0; c < d; ++c) {
    float a = hs[c] + ht[c];
    float p = hs[c] * ht[c];
    ew += a * ew_w[c] + p * ew_w[d + c];
    ep += a * ep_w[c] + p * ep_w[d + c];
  }
  ew += ew_b[0]; ep += ep_b[0];
  out_ew[e] = fmaxf(ew, 0.0f);
  out_ep[e] = ep;
}

// final 512->1 linear over bf16 activations (optional relu)
__global__ void k_final_dot(const __bf16* __restrict__ act, const float* __restrict__ w,
                            const float* __restrict__ b, float* __restrict__ out,
                            int M, int K, int relu) {
  int r = blockIdx.x * blockDim.x + threadIdx.x;
  if (r >= M) return;
  const __bf16* a = act + (size_t)r * K;
  float s = 0.0f;
  for (int k = 0; k < K; ++k) s += (float)a[k] * w[k];
  s += b[0];
  out[r] = relu ? fmaxf(s, 0.0f) : s;
}

// ---------------------------------------------------------------------------
// host side
// ---------------------------------------------------------------------------
static inline unsigned cdiv(size_t a, size_t b) { return (unsigned)((a + b - 1) / b); }

extern "C" void kernel_launch(void* const* d_in, const int* in_sizes, int n_in,
                              void* d_out, int out_size, void* d_ws, size_t ws_size,
                              hipStream_t stream) {
  (void)in_sizes; (void)n_in; (void)out_size; (void)ws_size;

  const float* x    = (const float*)d_in[IN_X];
  const int*   se   = (const int*)d_in[IN_SUP];
  const int*   me   = (const int*)d_in[IN_MSG];
  const float* mew  = (const float*)d_in[IN_MSGW];
  float* out = (float*)d_out;

  // ---- workspace carve-out (256B aligned) ----
  char* base = (char*)d_ws;
  size_t off = 0;
  auto alloc = [&](size_t bytes) -> char* {
    char* p = base + off;
    off += (bytes + 255) & ~(size_t)255;
    return p;
  };
  float*  agg   = (float*) alloc((size_t)N_NODES * HID * 4);
  float*  cnt   = (float*) alloc((size_t)N_NODES * 4);
  __bf16* xb    = (__bf16*)alloc((size_t)N_NODES * HID * 2);
  __bf16* aggb  = (__bf16*)alloc((size_t)N_NODES * HID * 2);
  float*  h1    = (float*) alloc((size_t)N_NODES * HID * 4);
  float*  h2    = (float*) alloc((size_t)N_NODES * HID * 4);
  __bf16* cw1n  = (__bf16*)alloc((size_t)HID * HID * 2);
  __bf16* cw1r  = (__bf16*)alloc((size_t)HID * HID * 2);
  __bf16* cw2n  = (__bf16*)alloc((size_t)HID * HID * 2);
  __bf16* cw2r  = (__bf16*)alloc((size_t)HID * HID * 2);
  __bf16* nf    = (__bf16*)alloc((size_t)E_SUP * 512 * 2);
  __bf16* W0t   = (__bf16*)alloc((size_t)2048 * 512 * 2);
  __bf16* W1t   = (__bf16*)alloc((size_t)1024 * 2048 * 2);
  __bf16* W2t   = (__bf16*)alloc((size_t)512 * 1024 * 2);
  __bf16* act1  = (__bf16*)alloc((size_t)E_SUP * 2048 * 2);
  __bf16* act2  = (__bf16*)alloc((size_t)E_SUP * 1024 * 2);
  __bf16* act3  = (__bf16*)alloc((size_t)E_SUP * 512 * 2);
  float*  stats = (float*) alloc((size_t)4096 * 4);   // [colsum(2048) | colsq(2048)]
  float*  colsum = stats;
  float*  colsq  = stats + 2048;

  const int TB = 256;

  auto gemm_blocks = [&](int M, int N) -> unsigned {
    return cdiv((size_t)(M / 16), 4) * (unsigned)(N / 128);
  };
  auto gemm_f32 = [&](const __bf16* A, const __bf16* Bt, float* C, const float* bias,
                      int M, int N, int K, bool accum) {
    unsigned blocks = gemm_blocks(M, N);
    if (accum)
      k_gemm_wmma<false, true><<<blocks, 128, 0, stream>>>(A, Bt, (void*)C, bias, M, N, K);
    else
      k_gemm_wmma<false, false><<<blocks, 128, 0, stream>>>(A, Bt, (void*)C, bias, M, N, K);
  };
  auto gemm_bf16 = [&](const __bf16* A, const __bf16* Bt, __bf16* C, const float* bias,
                       int M, int N, int K) {
    unsigned blocks = gemm_blocks(M, N);
    k_gemm_wmma<true, false><<<blocks, 128, 0, stream>>>(A, Bt, (void*)C, bias, M, N, K);
  };
  auto wconv = [&](int idx, __bf16* dst, int K, int N) {
    k_wt_convert<<<cdiv((size_t)K * N, TB), TB, 0, stream>>>((const float*)d_in[idx], dst, K, N);
  };

  // ===== GNN: conv weights to bf16 [N][K] =====
  wconv(IN_W1N, cw1n, IN_C, HID);
  wconv(IN_W1R, cw1r, IN_C, HID);
  wconv(IN_W2N, cw2n, HID, HID);
  wconv(IN_W2R, cw2r, HID, HID);

  // edge in-degree (shared by both convs)
  k_zero_f32<<<cdiv(N_NODES, TB), TB, 0, stream>>>(cnt, N_NODES);
  k_sage_cnt<<<cdiv(E_MSG, TB), TB, 0, stream>>>(me, cnt, E_MSG);

  // ---- conv1: h1 = mean_agg(x)@w1n + b1 + x@w1r ----
  k_zero_f32<<<cdiv((size_t)N_NODES * HID, TB), TB, 0, stream>>>(agg, (size_t)N_NODES * HID);
  k_sage_msg<<<cdiv((size_t)E_MSG * HID, TB), TB, 0, stream>>>(x, me, mew, agg, E_MSG, HID);
  k_sage_fin<<<cdiv((size_t)N_NODES * HID, TB), TB, 0, stream>>>(agg, cnt, aggb, N_NODES, HID);
  k_f32_to_bf16<<<cdiv((size_t)N_NODES * HID, TB), TB, 0, stream>>>(x, xb, (size_t)N_NODES * HID);
  gemm_f32(aggb, cw1n, h1, (const float*)d_in[IN_B1], N_NODES, HID, HID, false);
  gemm_f32(xb,   cw1r, h1, nullptr,                   N_NODES, HID, HID, true);

  // ---- conv2: h2 = mean_agg(h1)@w2n + b2 + h1@w2r ----
  k_zero_f32<<<cdiv((size_t)N_NODES * HID, TB), TB, 0, stream>>>(agg, (size_t)N_NODES * HID);
  k_sage_msg<<<cdiv((size_t)E_MSG * HID, TB), TB, 0, stream>>>(h1, me, mew, agg, E_MSG, HID);
  k_sage_fin<<<cdiv((size_t)N_NODES * HID, TB), TB, 0, stream>>>(agg, cnt, aggb, N_NODES, HID);
  k_f32_to_bf16<<<cdiv((size_t)N_NODES * HID, TB), TB, 0, stream>>>(h1, xb, (size_t)N_NODES * HID);
  gemm_f32(aggb, cw2n, h2, (const float*)d_in[IN_B2], N_NODES, HID, HID, false);
  gemm_f32(xb,   cw2r, h2, nullptr,                   N_NODES, HID, HID, true);

  // ---- GNN edge heads -> out[0:100K) = gnn_ew, out[100K:200K) = gnn_ep ----
  k_gnn_edge_head<<<cdiv(E_SUP, TB), TB, 0, stream>>>(
      h2, se, (const float*)d_in[IN_EWW], (const float*)d_in[IN_EWB],
      (const float*)d_in[IN_EPW], (const float*)d_in[IN_EPB],
      out + 0, out + E_SUP, E_SUP, HID);

  // ===== Node-only decoder =====
  k_build_nf<<<cdiv((size_t)E_SUP * IN_C, TB), TB, 0, stream>>>(x, se, nf, E_SUP, IN_C);

  auto run_mlp = [&](int bi, float* outp, int final_relu) {
    // weights -> bf16 transposed
    wconv(bi + 0, W0t, 512, 2048);
    wconv(bi + 4, W1t, 2048, 1024);
    wconv(bi + 8, W2t, 1024, 512);

    struct Layer { const __bf16* A; __bf16* Bout; const __bf16* Wt; int N, K; int gi; };
    Layer L[3] = {
      { nf,   act1, W0t, 2048, 512,  bi + 0 },
      { act1, act2, W1t, 1024, 2048, bi + 4 },
      { act2, act3, W2t, 512,  1024, bi + 8 },
    };
    for (int li = 0; li < 3; ++li) {
      gemm_bf16(L[li].A, L[li].Wt, L[li].Bout, (const float*)d_in[L[li].gi + 1],
                E_SUP, L[li].N, L[li].K);
      k_zero_f32<<<cdiv(4096, TB), TB, 0, stream>>>(stats, 4096);
      const int ROWS = 2048;
      dim3 sgrid(cdiv(L[li].N, TB), cdiv(E_SUP, ROWS));
      k_bn_stats<<<sgrid, TB, 0, stream>>>(L[li].Bout, colsum, colsq, E_SUP, L[li].N, ROWS);
      k_bn_apply<<<cdiv((size_t)E_SUP * L[li].N, TB), TB, 0, stream>>>(
          L[li].Bout, colsum, colsq,
          (const float*)d_in[L[li].gi + 2], (const float*)d_in[L[li].gi + 3],
          E_SUP, L[li].N);
    }
    k_final_dot<<<cdiv(E_SUP, TB), TB, 0, stream>>>(
        act3, (const float*)d_in[bi + 12], (const float*)d_in[bi + 13],
        outp, E_SUP, 512, final_relu);
  };

  // out[200K:300K) = nod_ew = relu(wt MLP); out[300K:400K) = nod_ep = prob MLP
  run_mlp(IN_PROB, out + 3 * E_SUP, /*relu=*/0);
  run_mlp(IN_WT,   out + 2 * E_SUP, /*relu=*/1);
}
